// GraphTransformerConv_8564164789034
// MI455X (gfx1250) — compile-verified, matrix operand order checked by reading, not run on hardware
//
#include <hip/hip_runtime.h>
#include <cstdint>
#include <cstddef>

#define NN    30000
#define INC   128
#define OUTC  64
#define HH    4
#define HD    256      // H*OUTC
#define MM    30
#define MP    32       // padded M
#define KG    10       // gumbel samples
#define EE    480000
#define EPSF  1e-6f

typedef __attribute__((ext_vector_type(16))) __bf16 v16bf;
typedef __attribute__((ext_vector_type(8)))  float  v8f;

// ---------------------------------------------------------------------------
// WMMA helpers (CDNA5 v_wmma_f32_16x16x32_bf16, wave32)
// A (16x32): lane m=lane&15 (half=lane>>4 shifts K by 8); VGPR j holds K pair
//   kb = (j<4 ? 2j : 16+2(j-4)) + 8*half  -> pairs contiguous in memory if the
//   16-row operand is row-major with pairs along K.
// B (32x16): symmetric with lanes holding columns; we store B = W^T so each
//   output column n reads weight row (col0+n) contiguously along K.
// C/D (16x16 f32): VGPR r, lanes 0-15 -> row r, lanes 16-31 -> row r+8.
// ---------------------------------------------------------------------------
__device__ __forceinline__ v16bf frag_load(const __bf16* base, int row_stride, int lane) {
  int r = lane & 15, half = lane >> 4;
  const __bf16* p = base + (size_t)r * row_stride + half * 8;
  v16bf f;
#pragma unroll
  for (int j = 0; j < 4; ++j) { f[2*j]   = p[2*j];      f[2*j+1] = p[2*j+1];      }
#pragma unroll
  for (int j = 0; j < 4; ++j) { f[8+2*j] = p[16+2*j];   f[9+2*j] = p[16+2*j+1];   }
  return f;
}

__device__ __forceinline__ v8f wmma_bf16(v16bf a, v16bf b, v8f c) {
  return __builtin_amdgcn_wmma_f32_16x16x32_bf16(false, a, false, b, (short)0, c, false, false);
}

// Async global->LDS copy of one 16B chunk per lane (gfx1250 ASYNCcnt path).
// VDST VGPR carries the wave-relative LDS byte offset (= low 32 bits of the
// generic pointer for LDS apertured addresses).
__device__ __forceinline__ void async_copy_b128(unsigned lds_off, const void* gptr) {
  unsigned long long ga = (unsigned long long)(uintptr_t)gptr;
  asm volatile("global_load_async_to_lds_b128 %0, %1, off"
               :: "v"(lds_off), "v"(ga) : "memory");
}
__device__ __forceinline__ void async_wait0() {
  asm volatile("s_wait_asynccnt 0x0" ::: "memory");
}

__device__ __forceinline__ void atomicMaxF(float* addr, float val) {
  int old = __float_as_int(*addr);
  while (__int_as_float(old) < val) {
    int prev = atomicCAS((int*)addr, old, __float_as_int(val));
    if (prev == old) break;
    old = prev;
  }
}

__device__ __forceinline__ float dot32(const float* a, const float* b) {
  float s = 0.f;
#pragma unroll
  for (int m = 0; m < MP; ++m) s += a[m] * b[m];
  return s;
}

// ---------------------------------------------------------------------------
// Stage 0: init accumulators
// ---------------------------------------------------------------------------
__global__ void k_init(float* kvsT, float* ks_sum, float* colsum,
                       float* din, float* dout, float* loss, float* maxk) {
  int i = blockIdx.x * blockDim.x + threadIdx.x;
  if (i < HH * KG * OUTC * MP) kvsT[i] = 0.f;
  if (i < HH * KG * MP)        ks_sum[i] = 0.f;
  if (i < HH * MP)             colsum[i] = 0.f;
  if (i < NN) { din[i] = 0.f; dout[i] = 0.f; }
  if (i == 0) loss[0] = 0.f;
  if (i < HH) maxk[i] = -3.4e38f;
}

// ---------------------------------------------------------------------------
// Stage 1: f32 -> bf16 conversions
// ---------------------------------------------------------------------------
__global__ void k_cvt(const float* src, __bf16* dst, int n) {
  int i = blockIdx.x * blockDim.x + threadIdx.x;
  if (i < n) dst[i] = (__bf16)src[i];
}

__global__ void k_cvt_wqkv(const float* Wq, const float* Wk, const float* Wv, __bf16* dst) {
  int i = blockIdx.x * blockDim.x + threadIdx.x;   // 768*128
  if (i >= 3 * HD * INC) return;
  int row = i >> 7, c = i & 127;
  float v = (row < HD) ? Wq[row * INC + c]
          : (row < 2 * HD) ? Wk[(row - HD) * INC + c]
          : Wv[(row - 2 * HD) * INC + c];
  dst[i] = (__bf16)v;
}

// ---------------------------------------------------------------------------
// Stage 2: fused q/k/v GEMM  [N,128] x [128,768] via WMMA bf16
// grid (N/16, 12), block 128 (4 waves, one 16x16 tile each).
// The shared 16x128 A tile (4 KB) is staged into LDS with async copies so the
// 4 warps (and their 4 wmma K-steps) read A from LDS instead of 4x from L2.
// ---------------------------------------------------------------------------
__global__ void k_gemm_qkv(const __bf16* zb, const __bf16* wb,
                           const float* bq, const float* bk, const float* bv,
                           float* qkv) {
  __shared__ __bf16 sA[16 * INC];                       // 4 KB
  int tid  = threadIdx.x;
  int lane = tid & 31, warp = tid >> 5;
  int row0 = blockIdx.x * 16;
  int col0 = (blockIdx.y * 4 + warp) * 16;

  {  // stage A tile: 256 chunks of 16B, 2 per thread
    const char* gbase = (const char*)(zb + (size_t)row0 * INC);
    unsigned lds0 = (unsigned)(uintptr_t)(&sA[0]);
#pragma unroll
    for (int it = 0; it < 2; ++it) {
      int chunk = tid + it * 128;
      async_copy_b128(lds0 + chunk * 16, gbase + chunk * 16);
    }
    async_wait0();
  }
  __syncthreads();

  v8f c = {0.f, 0.f, 0.f, 0.f, 0.f, 0.f, 0.f, 0.f};
#pragma unroll
  for (int kk = 0; kk < INC; kk += 32) {
    v16bf a = frag_load(sA + kk, INC, lane);
    v16bf b = frag_load(wb + (size_t)col0 * INC + kk, INC, lane);
    c = wmma_bf16(a, b, c);
  }
  int half = lane >> 4, nc = lane & 15;
  int col = col0 + nc, seg = col >> 8, cc = col & 255;
  float bias = (seg == 0) ? bq[cc] : (seg == 1) ? bk[cc] : bv[cc];
  float* dst = qkv + (size_t)seg * NN * HD;
#pragma unroll
  for (int r = 0; r < 8; ++r) {
    int row = row0 + r + half * 8;
    dst[(size_t)row * HD + cc] = c[r] + bias;
  }
}

// ---------------------------------------------------------------------------
// Stage 3: Performer feature maps
// ---------------------------------------------------------------------------
// scale = tau^-0.5 * d^-0.25 = 2 * 64^-0.25 = 0.70710678
#define FSCALE 0.70710678f
#define RATIO  0.18257419f   // 30^-0.5

__global__ void k_qfeat(const float* q, const float* proj, float* qp, __bf16* qpb) {
  __shared__ float sproj[MM * OUTC];
  for (int i = threadIdx.x; i < MM * OUTC; i += blockDim.x) sproj[i] = proj[i];
  __syncthreads();
  int t = blockIdx.x * blockDim.x + threadIdx.x;
  if (t >= NN * HH) return;
  int n = t >> 2, h = t & 3;
  const float* qr = q + (size_t)n * HD + h * OUTC;
  float diag = 0.f;
  for (int d = 0; d < OUTC; ++d) { float x = qr[d] * FSCALE; diag += x * x; }
  diag *= 0.5f;
  float dd[MM]; float mx = -3.4e38f;
  for (int m = 0; m < MM; ++m) {
    float s = 0.f;
    for (int d = 0; d < OUTC; ++d) s += (qr[d] * FSCALE) * sproj[m * OUTC + d];
    dd[m] = s; mx = fmaxf(mx, s);
  }
  float* o = qp + (size_t)t * MP; __bf16* ob = qpb + (size_t)t * MP;
  for (int m = 0; m < MM; ++m) {
    float v = RATIO * (expf(dd[m] - diag - mx) + EPSF);
    o[m] = v; ob[m] = (__bf16)v;
  }
  o[30] = 0.f; o[31] = 0.f; ob[30] = (__bf16)0.f; ob[31] = (__bf16)0.f;
}

__global__ void k_kfeat1(const float* k, const float* proj,
                         float* ddk, float* diagk, float* maxk) {
  __shared__ float sproj[MM * OUTC];
  for (int i = threadIdx.x; i < MM * OUTC; i += blockDim.x) sproj[i] = proj[i];
  __syncthreads();
  int t = blockIdx.x * blockDim.x + threadIdx.x;
  if (t >= NN * HH) return;
  int n = t >> 2, h = t & 3;
  const float* kr = k + (size_t)n * HD + h * OUTC;
  float diag = 0.f;
  for (int d = 0; d < OUTC; ++d) { float x = kr[d] * FSCALE; diag += x * x; }
  diagk[t] = diag * 0.5f;
  float mx = -3.4e38f;
  for (int m = 0; m < MM; ++m) {
    float s = 0.f;
    for (int d = 0; d < OUTC; ++d) s += (kr[d] * FSCALE) * sproj[m * OUTC + d];
    ddk[(size_t)t * MP + m] = s; mx = fmaxf(mx, s);
  }
  atomicMaxF(&maxk[h], mx);
}

__global__ void k_kfeat2(const float* ddk, const float* diagk, const float* maxk,
                         float* kp, float* colsum) {
  int t = blockIdx.x * blockDim.x + threadIdx.x;
  if (t >= NN * HH) return;
  int h = t & 3;
  float mx = maxk[h], dg = diagk[t];
  float* o = kp + (size_t)t * MP;
  for (int m = 0; m < MM; ++m) {
    float v = RATIO * (expf(ddk[(size_t)t * MP + m] - dg - mx) + EPSF);
    o[m] = v;
    atomicAdd(&colsum[h * MP + m], v);
  }
  o[30] = 0.f; o[31] = 0.f;
}

// ---------------------------------------------------------------------------
// Stage 4: kvs[h,k] = sum_n (kp[n,h,:]*g[n,h,k]) (outer) v[n,h,:]; ks_sum too.
// Stored transposed: kvsT[hk][d][m] so z_num's B-fragments are contiguous.
// grid (30 chunks, 40 hk), block 256: thread = (m = t&31, dgroup = t>>5 -> 8 d)
// ---------------------------------------------------------------------------
__global__ void k_kvs(const float* kp, const float* gum, const float* v,
                      float* kvsT, float* ks_sum) {
  int hk = blockIdx.y, h = hk / KG, kq = hk % KG;
  int t = threadIdx.x, m = t & 31, d0 = (t >> 5) * 8;
  int n0 = blockIdx.x * 1000, n1 = n0 + 1000;
  float acc[8] = {0.f, 0.f, 0.f, 0.f, 0.f, 0.f, 0.f, 0.f};
  float ksa = 0.f;
  for (int n = n0; n < n1; ++n) {
    float g  = expf(gum[((size_t)n * HH + h) * KG + kq]);
    float kg = kp[((size_t)n * HH + h) * MP + m] * g;
    const float* vr = v + (size_t)n * HD + h * OUTC + d0;
#pragma unroll
    for (int d = 0; d < 8; ++d) acc[d] += kg * vr[d];
    if (d0 == 0) ksa += kg;
  }
#pragma unroll
  for (int d = 0; d < 8; ++d)
    atomicAdd(&kvsT[((size_t)hk * OUTC + d0 + d) * MP + m], acc[d]);
  if (d0 == 0) atomicAdd(&ks_sum[hk * MP + m], ksa);
}

__global__ void k_zden(const float* qp, const float* ks_sum, float* zden) {
  int i = blockIdx.x * blockDim.x + threadIdx.x;     // (n,h,k)
  if (i >= NN * HH * KG) return;
  int nh = i / KG, hk = (nh & 3) * KG + (i % KG);
  zden[i] = dot32(qp + (size_t)nh * MP, ks_sum + hk * MP);
}

// ---------------------------------------------------------------------------
// Stage 5: z_out[n,h,d] = (1/K) * sum_k (qp[n,h,:] . kvsT[h,k,:,d]) / zden[n,h,k]
// WMMA: A = qp tile [16 nodes x 32 m], B = kvsT [32 m x 16 d]; 10 wmma per tile
// grid (N/16, H), block 128 (warp = d-tile of 16)
// ---------------------------------------------------------------------------
__global__ void k_znum(const __bf16* qpb, const __bf16* kvb,
                       const float* zden, float* zout) {
  int lane = threadIdx.x & 31, warp = threadIdx.x >> 5;
  int row0 = blockIdx.x * 16, h = blockIdx.y, d0 = warp * 16;
  v16bf a = frag_load(qpb + ((size_t)row0 * HH + h) * MP, HH * MP, lane);
  int half = lane >> 4, nc = lane & 15;
  float accf[8] = {0.f, 0.f, 0.f, 0.f, 0.f, 0.f, 0.f, 0.f};
  for (int kq = 0; kq < KG; ++kq) {
    int hk = h * KG + kq;
    v16bf b = frag_load(kvb + ((size_t)hk * OUTC + d0) * MP, MP, lane);
    v8f c = {0.f, 0.f, 0.f, 0.f, 0.f, 0.f, 0.f, 0.f};
    c = wmma_bf16(a, b, c);
#pragma unroll
    for (int r = 0; r < 8; ++r) {
      int row = row0 + r + half * 8;
      float den = zden[((size_t)row * HH + h) * KG + kq];
      accf[r] += c[r] / den;
    }
  }
#pragma unroll
  for (int r = 0; r < 8; ++r) {
    int row = row0 + r + half * 8;
    zout[((size_t)row * HH + h) * OUTC + d0 + nc] = accf[r] * (1.0f / KG);
  }
}

// ---------------------------------------------------------------------------
// Stage 6: edges
// ---------------------------------------------------------------------------
__global__ void k_deg(const int* ei, float* din, float* dout) {
  int e = blockIdx.x * blockDim.x + threadIdx.x;
  if (e >= EE) return;
  atomicAdd(&dout[ei[e]], 1.f);        // row = start
  atomicAdd(&din[ei[EE + e]], 1.f);    // col = end
}

__global__ void k_norm(const float* qp, const float* colsum, float* normz) {
  int t = blockIdx.x * blockDim.x + threadIdx.x;
  if (t >= NN * HH) return;
  normz[t] = dot32(qp + (size_t)t * MP, colsum + (t & 3) * MP);
}

__global__ void k_loss(const int* ei, const float* qp, const float* kp,
                       const float* normz, const float* din, float* loss) {
  __shared__ float red[256];
  int e = blockIdx.x * 256 + threadIdx.x;
  float acc = 0.f;
  if (e < EE) {
    int s = ei[e], dnode = ei[EE + e];
    float w = 1.f / din[dnode];
    for (int h = 0; h < HH; ++h) {
      float num = dot32(qp + ((size_t)dnode * HH + h) * MP,
                        kp + ((size_t)s * HH + h) * MP);
      acc += logf(num / normz[dnode * HH + h]);
    }
    acc *= w;
  }
  red[threadIdx.x] = acc;
  __syncthreads();
  for (int off = 128; off; off >>= 1) {
    if (threadIdx.x < off) red[threadIdx.x] += red[threadIdx.x + off];
    __syncthreads();
  }
  if (threadIdx.x == 0) atomicAdd(loss, red[0]);
}

__global__ void k_conv(const int* ei, const float* din, const float* dout,
                       const float* brb, const float* v, float* zout) {
  int i = blockIdx.x * blockDim.x + threadIdx.x;   // (e,h)
  if (i >= EE * HH) return;
  int e = i >> 2, h = i & 3;
  int s = ei[e], c = ei[EE + e];
  float dn = rsqrtf(din[c]) * rsqrtf(dout[s]);
  float bh = 1.f / (1.f + expf(-brb[h]));
  float coef = dn * bh;
  const float* vr = v + (size_t)s * HD + h * OUTC;
  float* zo = zout + ((size_t)c * HH + h) * OUTC;
  for (int d = 0; d < OUTC; ++d) atomicAdd(&zo[d], coef * vr[d]);
}

// ---------------------------------------------------------------------------
// Stage 7: output GEMM [N,256] x [256,64] via WMMA bf16; grid (N/16,1) block 128
// A tile (16x256 bf16 = 8 KB) staged into LDS via async copies, shared by warps.
// ---------------------------------------------------------------------------
__global__ void k_gemm_out(const __bf16* zob, const __bf16* wob,
                           const float* bo, float* out) {
  __shared__ __bf16 sA[16 * HD];                        // 8 KB
  int tid = threadIdx.x;
  int lane = tid & 31, warp = tid >> 5;
  int row0 = blockIdx.x * 16, col0 = warp * 16;

  {  // stage A tile: 512 chunks of 16B, 4 per thread
    const char* gbase = (const char*)(zob + (size_t)row0 * HD);
    unsigned lds0 = (unsigned)(uintptr_t)(&sA[0]);
#pragma unroll
    for (int it = 0; it < 4; ++it) {
      int chunk = tid + it * 128;
      async_copy_b128(lds0 + chunk * 16, gbase + chunk * 16);
    }
    async_wait0();
  }
  __syncthreads();

  v8f c = {0.f, 0.f, 0.f, 0.f, 0.f, 0.f, 0.f, 0.f};
#pragma unroll
  for (int kk = 0; kk < HD; kk += 32) {
    v16bf a = frag_load(sA + kk, HD, lane);
    v16bf b = frag_load(wob + (size_t)col0 * HD + kk, HD, lane);
    c = wmma_bf16(a, b, c);
  }
  int half = lane >> 4, nc = lane & 15;
  float bias = bo[col0 + nc];
#pragma unroll
  for (int r = 0; r < 8; ++r)
    out[(size_t)(row0 + r + half * 8) * OUTC + col0 + nc] = c[r] + bias;
}

__global__ void k_final_loss(const float* loss, float* out) {
  out[(size_t)NN * OUTC] = loss[0] / (float)(EE * HH);
}

// ---------------------------------------------------------------------------
extern "C" void kernel_launch(void* const* d_in, const int* in_sizes, int n_in,
                              void* d_out, int out_size, void* d_ws, size_t ws_size,
                              hipStream_t stream) {
  const float* z    = (const float*)d_in[0];
  const int*   ei   = (const int*)  d_in[1];
  const float* gum  = (const float*)d_in[2];
  const float* proj = (const float*)d_in[3];
  const float* Wq   = (const float*)d_in[4];
  const float* bq   = (const float*)d_in[5];
  const float* Wk   = (const float*)d_in[6];
  const float* bk   = (const float*)d_in[7];
  const float* Wv   = (const float*)d_in[8];
  const float* bv   = (const float*)d_in[9];
  const float* Wo   = (const float*)d_in[10];
  const float* bo   = (const float*)d_in[11];
  const float* brb  = (const float*)d_in[12];
  float* out = (float*)d_out;

  // bump allocator over d_ws
  char* w = (char*)d_ws;
  auto alloc = [&](size_t bytes) {
    void* p = (void*)w;
    w += (bytes + 255) & ~(size_t)255;
    return p;
  };
  __bf16* zb     = (__bf16*)alloc((size_t)NN * INC * 2);         // 7.68 MB
  __bf16* wb     = (__bf16*)alloc((size_t)3 * HD * INC * 2);     // 196 KB
  __bf16* wob    = (__bf16*)alloc((size_t)OUTC * HD * 2);        // 32 KB
  float*  qkv    = (float*) alloc((size_t)3 * NN * HD * 4);      // 92.16 MB
  float*  qp     = (float*) alloc((size_t)NN * HH * MP * 4);     // 15.36 MB
  __bf16* qpb    = (__bf16*)alloc((size_t)NN * HH * MP * 2);     // 7.68 MB
  float*  kp     = (float*) alloc((size_t)NN * HH * MP * 4);     // 15.36 MB
  float*  kvsT   = (float*) alloc((size_t)HH * KG * OUTC * MP * 4);
  __bf16* kvb    = (__bf16*)alloc((size_t)HH * KG * OUTC * MP * 2);
  float*  ks_sum = (float*) alloc((size_t)HH * KG * MP * 4);
  float*  zden   = (float*) alloc((size_t)NN * HH * KG * 4);     // 4.8 MB
  float*  zout   = (float*) alloc((size_t)NN * HH * OUTC * 4);   // 30.72 MB
  float*  colsum = (float*) alloc((size_t)HH * MP * 4);
  float*  maxk   = (float*) alloc((size_t)HH * 4);
  float*  din    = (float*) alloc((size_t)NN * 4);
  float*  dout_  = (float*) alloc((size_t)NN * 4);
  float*  normz  = (float*) alloc((size_t)NN * HH * 4);
  float*  loss   = (float*) alloc(4);
  // aliases into dead slabs:
  float*  q      = qkv;
  float*  kmat   = qkv + (size_t)NN * HD;
  float*  v      = qkv + (size_t)2 * NN * HD;
  float*  ddk    = qkv;                               // q dead after k_qfeat
  float*  diagk  = qkv + (size_t)NN * HH * MP;
  __bf16* zob    = (__bf16*)kmat;                     // k dead after k_kfeat1

  dim3 B256(256);
  // 0: init accumulators
  k_init<<<dim3(320), B256, 0, stream>>>(kvsT, ks_sum, colsum, din, dout_, loss, maxk);
  // 1: conversions
  k_cvt<<<dim3((NN * INC + 255) / 256), B256, 0, stream>>>(z, zb, NN * INC);
  k_cvt_wqkv<<<dim3((3 * HD * INC + 255) / 256), B256, 0, stream>>>(Wq, Wk, Wv, wb);
  k_cvt<<<dim3((OUTC * HD + 255) / 256), B256, 0, stream>>>(Wo, wob, OUTC * HD);
  // 2: q/k/v GEMM (WMMA + async LDS staging)
  k_gemm_qkv<<<dim3(NN / 16, 12), dim3(128), 0, stream>>>(zb, wb, bq, bk, bv, qkv);
  // 3: feature maps
  k_qfeat<<<dim3((NN * HH + 255) / 256), B256, 0, stream>>>(q, proj, qp, qpb);
  k_kfeat1<<<dim3((NN * HH + 255) / 256), B256, 0, stream>>>(kmat, proj, ddk, diagk, maxk);
  k_kfeat2<<<dim3((NN * HH + 255) / 256), B256, 0, stream>>>(ddk, diagk, maxk, kp, colsum);
  // 4: kvs / ks_sum / zden
  k_kvs<<<dim3(30, HH * KG), B256, 0, stream>>>(kp, gum, v, kvsT, ks_sum);
  k_cvt<<<dim3((HH * KG * OUTC * MP + 255) / 256), B256, 0, stream>>>(kvsT, kvb, HH * KG * OUTC * MP);
  k_zden<<<dim3((NN * HH * KG + 255) / 256), B256, 0, stream>>>(qp, ks_sum, zden);
  // 5: z_num -> z_out (WMMA)
  k_znum<<<dim3(NN / 16, HH), dim3(128), 0, stream>>>(qpb, kvb, zden, zout);
  // 6: edges
  k_deg<<<dim3(EE / 256), B256, 0, stream>>>(ei, din, dout_);
  k_norm<<<dim3((NN * HH + 255) / 256), B256, 0, stream>>>(qp, colsum, normz);
  k_loss<<<dim3(EE / 256), B256, 0, stream>>>(ei, qp, kp, normz, din, loss);
  k_conv<<<dim3(EE * HH / 256), B256, 0, stream>>>(ei, din, dout_, brb, v, zout);
  // 7: output GEMM (WMMA + async LDS staging) + loss
  k_cvt<<<dim3((NN * HD + 255) / 256), B256, 0, stream>>>(zout, zob, NN * HD);
  k_gemm_out<<<dim3(NN / 16, 1), dim3(128), 0, stream>>>(zob, wob, bo, out);
  k_final_loss<<<dim3(1), dim3(1), 0, stream>>>(loss, out);

  (void)in_sizes; (void)n_in; (void)out_size; (void)ws_size;
}
